// SAGE_54391465836676
// MI455X (gfx1250) — matrix operand order repaired and checked by reading, not compiled, not run on hardware
//
#include <hip/hip_runtime.h>
#include <hip/hip_bf16.h>
#include <stdint.h>

typedef __attribute__((ext_vector_type(16))) __bf16 v16bf;
typedef __attribute__((ext_vector_type(8)))  float  v8f;

#define SAGE_N1 200000
#define SAGE_N2 50000
#define SAGE_N3 10000
#define SAGE_E1 800000
#define SAGE_E2 160000
#define SAGE_D  256      // D_IN == D_HID
#define SAGE_DO 128      // D_OUT

union sage_frag { v16bf v; uint4 q[2]; };

__device__ __forceinline__ uint32_t sage_pk2(__bf16 a, __bf16 b) {
  union { __bf16 h[2]; uint32_t u; } c;
  c.h[0] = a; c.h[1] = b;
  return c.u;
}

// ---------------------------------------------------------------- zero
__global__ void sage_zero_kernel(float* __restrict__ p, long n) {
  long i = (long)blockIdx.x * blockDim.x + threadIdx.x;
  long stride = (long)gridDim.x * blockDim.x;
  for (; i < n; i += stride) p[i] = 0.0f;
}

// ------------------------------------------------------- edge scatter-add
// One wave per edge; each lane handles 8 contiguous features (256/32).
__global__ void sage_scatter_kernel(const float* __restrict__ feat,
                                    const int*   __restrict__ esrc,
                                    const int*   __restrict__ edst,
                                    float* __restrict__ agg,
                                    float* __restrict__ deg,
                                    int nedges) {
  int lane = threadIdx.x & 31;
  int wid  = (int)(((long)blockIdx.x * blockDim.x + threadIdx.x) >> 5);
  if (wid >= nedges) return;
  int s = esrc[wid];
  int d = edst[wid];
  const float4* sp = (const float4*)(feat + (size_t)s * SAGE_D) + lane * 2;
  float* dp = agg + (size_t)d * SAGE_D + lane * 8;
  float4 v0 = sp[0];
  float4 v1 = sp[1];
  atomicAdd(dp + 0, v0.x); atomicAdd(dp + 1, v0.y);
  atomicAdd(dp + 2, v0.z); atomicAdd(dp + 3, v0.w);
  atomicAdd(dp + 4, v1.x); atomicAdd(dp + 5, v1.y);
  atomicAdd(dp + 6, v1.z); atomicAdd(dp + 7, v1.w);
  if (lane == 0) atomicAdd(deg + d, 1.0f);
}

// ---------------------------------------- f32 -> bf16 hi/lo split (2/thread)
__global__ void sage_cvt_kernel(const float* __restrict__ src,
                                uint32_t* __restrict__ dstH,
                                uint32_t* __restrict__ dstL,
                                long npairs) {
  long j = (long)blockIdx.x * blockDim.x + threadIdx.x;
  long stride = (long)gridDim.x * blockDim.x;
  for (; j < npairs; j += stride) {
    float f0 = src[2 * j];
    float f1 = src[2 * j + 1];
    __bf16 h0 = (__bf16)f0, h1 = (__bf16)f1;
    dstH[j] = sage_pk2(h0, h1);
    dstL[j] = sage_pk2((__bf16)(f0 - (float)h0), (__bf16)(f1 - (float)h1));
  }
}

// -------------------------- mean normalize + bf16 hi/lo split (2/thread)
__global__ void sage_norm_kernel(const float* __restrict__ agg,
                                 const float* __restrict__ deg,
                                 uint32_t* __restrict__ hnH,
                                 uint32_t* __restrict__ hnL,
                                 int rows) {
  long npairs = (long)rows * (SAGE_D / 2);
  long j = (long)blockIdx.x * blockDim.x + threadIdx.x;
  long stride = (long)gridDim.x * blockDim.x;
  for (; j < npairs; j += stride) {
    long e0 = 2 * j;
    float dv = fmaxf(deg[e0 >> 8], 1.0f);      // D == 256; pair stays in-row
    float f0 = agg[e0] / dv;
    float f1 = agg[e0 + 1] / dv;
    __bf16 h0 = (__bf16)f0, h1 = (__bf16)f1;
    hnH[j] = sage_pk2(h0, h1);
    hnL[j] = sage_pk2((__bf16)(f0 - (float)h0), (__bf16)(f1 - (float)h1));
  }
}

// ---------------- pack [Wself;Wneigh] (512 x N) into B-fragment layout
// dword index = ((kk*2 + h)*N + col)*8 + i  holds bf16{K=kk*32+8h+2i+(i>=4?8:0), +1} @ col
__global__ void sage_packw_kernel(const float* __restrict__ Wself,
                                  const float* __restrict__ Wneigh,
                                  uint32_t* __restrict__ pkH,
                                  uint32_t* __restrict__ pkL,
                                  int N) {
  int idx = blockIdx.x * blockDim.x + threadIdx.x;
  int total = 16 * 2 * N * 8;
  if (idx >= total) return;
  int i   = idx & 7;
  int col = (idx >> 3) % N;
  int hh  = (idx >> 3) / N;          // 0..31
  int h   = hh & 1;
  int kk  = hh >> 1;
  int k   = kk * 32 + 8 * h + 2 * i + ((i >= 4) ? 8 : 0);   // 0..510, even
  const float* W = (k < SAGE_D) ? Wself : Wneigh;
  int krow = k & (SAGE_D - 1);
  float f0 = W[(size_t)krow * N + col];
  float f1 = W[(size_t)(krow + 1) * N + col];
  __bf16 h0 = (__bf16)f0, h1 = (__bf16)f1;
  pkH[idx] = sage_pk2(h0, h1);
  pkL[idx] = sage_pk2((__bf16)(f0 - (float)h0), (__bf16)(f1 - (float)h1));
}

// ------------------------------------------------ fused dual-GEMM + bias
// out = act( [Aself | Aneigh] (M x 512, bf16 hi+lo) @ packed W (512 x N) + b )
// Wave computes a 16x64 strip; split-bf16: acc += ah*bh + ah*bl + al*bh.
// Inner loop is pure b128 loads + 12 WMMAs per K-step (no conversions).
__global__ void __launch_bounds__(256)
sage_gemm_kernel(const __bf16* __restrict__ aSelfH, const __bf16* __restrict__ aSelfL,
                 const __bf16* __restrict__ aNeighH, const __bf16* __restrict__ aNeighL,
                 const uint32_t* __restrict__ WpkH, const uint32_t* __restrict__ WpkL,
                 const float* __restrict__ bias,
                 float* __restrict__ outF,
                 __bf16* __restrict__ outH, __bf16* __restrict__ outL, int outHRows,
                 int M, int N, int relu) {
  int lane   = threadIdx.x & 31;
  int wid    = (int)(((long)blockIdx.x * blockDim.x + threadIdx.x) >> 5);
  int tilesN = N >> 6;                    // 64-wide wave strips
  int tileM  = wid / tilesN;              // wave-uniform
  if (tileM * 16 >= M) return;            // uniform exit: EXEC stays all-1s
  int n0   = (wid % tilesN) << 6;
  int m0   = tileM << 4;
  int h    = lane >> 4;
  int colb = lane & 15;

  const __bf16* aSH = aSelfH  + (size_t)(m0 + colb) * SAGE_D + 8 * h;
  const __bf16* aSL = aSelfL  + (size_t)(m0 + colb) * SAGE_D + 8 * h;
  const __bf16* aNH = aNeighH + (size_t)(m0 + colb) * SAGE_D + 8 * h;
  const __bf16* aNL = aNeighL + (size_t)(m0 + colb) * SAGE_D + 8 * h;

  v8f acc0 = {}, acc1 = {}, acc2 = {}, acc3 = {};

#pragma unroll 1
  for (int kk = 0; kk < 16; ++kk) {       // K = 512 total, 32 per step
    const __bf16 *AH, *AL;
    int k0;
    if (kk < 8) { AH = aSH; AL = aSL; k0 = kk * 32; }
    else        { AH = aNH; AL = aNL; k0 = (kk - 8) * 32; }

    // A fragment: elems 0..7 <- K = k0+8h..+7 ; elems 8..15 <- K = k0+16+8h..+7
    sage_frag ah, al;
    {
      const uint4* p = (const uint4*)(AH + k0);
      ah.q[0] = p[0]; ah.q[1] = p[2];
      const uint4* q = (const uint4*)(AL + k0);
      al.q[0] = q[0]; al.q[1] = q[2];
    }

    size_t brow = ((size_t)(kk * 2 + h) * N) * 8;
#define SAGE_DO_TILE(t, acc)                                                     \
    {                                                                            \
      const uint4* pbh = (const uint4*)(WpkH + brow + (size_t)(n0 + 16*(t) + colb) * 8); \
      const uint4* pbl = (const uint4*)(WpkL + brow + (size_t)(n0 + 16*(t) + colb) * 8); \
      sage_frag bh, bl;                                                          \
      bh.q[0] = pbh[0]; bh.q[1] = pbh[1];                                        \
      bl.q[0] = pbl[0]; bl.q[1] = pbl[1];                                        \
      acc = __builtin_amdgcn_wmma_f32_16x16x32_bf16(false, ah.v, false, bh.v,    \
                                                    (short)0, acc, false, false);\
      acc = __builtin_amdgcn_wmma_f32_16x16x32_bf16(false, ah.v, false, bl.v,    \
                                                    (short)0, acc, false, false);\
      acc = __builtin_amdgcn_wmma_f32_16x16x32_bf16(false, al.v, false, bh.v,    \
                                                    (short)0, acc, false, false);\
    }
    SAGE_DO_TILE(0, acc0)
    SAGE_DO_TILE(1, acc1)
    SAGE_DO_TILE(2, acc2)
    SAGE_DO_TILE(3, acc3)
#undef SAGE_DO_TILE
  }

  // Epilogue: C/D layout -> row = m0 + 8h + i, col = n0 + 16t + (lane&15)
#pragma unroll
  for (int t = 0; t < 4; ++t) {
    v8f acc = (t == 0) ? acc0 : (t == 1) ? acc1 : (t == 2) ? acc2 : acc3;
    int col = n0 + 16 * t + colb;
    float bv = bias[col];
#pragma unroll
    for (int i = 0; i < 8; ++i) {
      int row = m0 + 8 * h + i;
      float v = acc[i] + bv;
      if (relu) v = fmaxf(v, 0.0f);
      outF[(size_t)row * N + col] = v;
      if (outH != nullptr && row < outHRows) {  // fused bf16 split of h1[:N3]
        __bf16 hb = (__bf16)v;
        outH[(size_t)row * N + col] = hb;
        outL[(size_t)row * N + col] = (__bf16)(v - (float)hb);
      }
    }
  }
}

// ----------------------------------------------------------------- launch
extern "C" void kernel_launch(void* const* d_in, const int* in_sizes, int n_in,
                              void* d_out, int out_size, void* d_ws, size_t ws_size,
                              hipStream_t stream) {
  const float* x   = (const float*)d_in[0];
  const float* Ws1 = (const float*)d_in[1];
  const float* Wn1 = (const float*)d_in[2];
  const float* b1  = (const float*)d_in[3];
  const float* Ws2 = (const float*)d_in[4];
  const float* Wn2 = (const float*)d_in[5];
  const float* b2  = (const float*)d_in[6];
  const int* e1s = (const int*)d_in[7];
  const int* e1d = (const int*)d_in[8];
  const int* e2s = (const int*)d_in[9];
  const int* e2d = (const int*)d_in[10];
  float* out = (float*)d_out;

  // ---- workspace layout ----
  const size_t NF1 = (size_t)SAGE_N2 * SAGE_D;   // 12.8M
  const size_t NF2 = (size_t)SAGE_N3 * SAGE_D;   // 2.56M
  float* agg1 = (float*)d_ws;                    // 12.8M f32   [layer1]
  float* deg1 = agg1 + NF1;                      // 50k  f32
  float* h1   = deg1 + SAGE_N2;                  // 12.8M f32
  __bf16* xh   = (__bf16*)(h1 + NF1);            // 12.8M bf16
  __bf16* xl   = xh + NF1;
  __bf16* hn1h = xl + NF1;
  __bf16* hn1l = hn1h + NF1;
  __bf16* h1h  = hn1l + NF1;                     // 2.56M bf16 (rows < N3)
  __bf16* h1l  = h1h + NF2;
  uint32_t* W1h = (uint32_t*)(h1l + NF2);        // 65536 u32
  uint32_t* W1l = W1h + 16 * 2 * SAGE_D  * 8 / 2; // = +32768
  uint32_t* W2h = W1l + 16 * 2 * SAGE_D  * 8 / 2;
  uint32_t* W2l = W2h + 16 * 2 * SAGE_DO * 8 / 2; // = +16384
  // layer-2 scratch aliases the (dead after normalize1) agg1 region:
  float* agg2 = agg1;                            // 2.56M f32
  float* deg2 = agg1 + NF2;                      // 10k  f32
  uint32_t* hn2h = (uint32_t*)(deg2 + SAGE_N3);  // 2.56M bf16 (as 1.28M u32)
  uint32_t* hn2l = hn2h + NF2 / 2;

  // ---- layer 1 ----
  sage_zero_kernel<<<2048, 256, 0, stream>>>(agg1, (long)(NF1 + SAGE_N2));
  {
    int blocks = (int)(((long)SAGE_E1 * 32 + 255) / 256);
    sage_scatter_kernel<<<blocks, 256, 0, stream>>>(x, e1s, e1d, agg1, deg1, SAGE_E1);
  }
  sage_cvt_kernel<<<2048, 256, 0, stream>>>(x, (uint32_t*)xh, (uint32_t*)xl,
                                            (long)(NF1 / 2));
  sage_packw_kernel<<<(16*2*SAGE_D*8 + 255)/256, 256, 0, stream>>>(Ws1, Wn1, W1h, W1l, SAGE_D);
  sage_packw_kernel<<<(16*2*SAGE_DO*8 + 255)/256, 256, 0, stream>>>(Ws2, Wn2, W2h, W2l, SAGE_DO);
  sage_norm_kernel<<<2048, 256, 0, stream>>>(agg1, deg1,
                                             (uint32_t*)hn1h, (uint32_t*)hn1l, SAGE_N2);
  {
    int waves  = (SAGE_N2 / 16) * (SAGE_D / 64);
    int blocks = (waves * 32 + 255) / 256;
    sage_gemm_kernel<<<blocks, 256, 0, stream>>>(xh, xl, hn1h, hn1l, W1h, W1l, b1,
                                                 h1, h1h, h1l, SAGE_N3,
                                                 SAGE_N2, SAGE_D, /*relu=*/1);
  }

  // ---- layer 2 ----
  sage_zero_kernel<<<1024, 256, 0, stream>>>(agg2, (long)(NF2 + SAGE_N3));
  {
    int blocks = (int)(((long)SAGE_E2 * 32 + 255) / 256);
    sage_scatter_kernel<<<blocks, 256, 0, stream>>>(h1, e2s, e2d, agg2, deg2, SAGE_E2);
  }
  sage_norm_kernel<<<1024, 256, 0, stream>>>(agg2, deg2, hn2h, hn2l, SAGE_N3);
  {
    int waves  = (SAGE_N3 / 16) * (SAGE_DO / 64);
    int blocks = (waves * 32 + 255) / 256;
    sage_gemm_kernel<<<blocks, 256, 0, stream>>>(h1h, h1l, (const __bf16*)hn2h,
                                                 (const __bf16*)hn2l, W2h, W2l, b2,
                                                 out, nullptr, nullptr, 0,
                                                 SAGE_N3, SAGE_DO, /*relu=*/0);
  }
}
